// OneStepModel_73478300500356
// MI455X (gfx1250) — matrix operand order repaired and checked by reading, not compiled, not run on hardware
//
#include <hip/hip_runtime.h>

// GRU language model forward for MI455X (gfx1250, wave32, WMMA bf16 path).
//
//   eg      = emb @ w_in + b_in            (exact fp32, 128x3072, L2-resident)
//   scan:   rg = h @ w_rec  (bf16 WMMA, fp32 accum) + gates in fp32
//   logits  = hs @ w_out + b_out           (bf16 WMMA, fp32 accum)
//
// Fragment layouts as packed here (wave32; A and B share contiguous k = 16*(lane>>4) + elem):
//   A 16x32 bf16 : lane = (m&15) + 16*(k>>4),     elem = k & 15
//   B 32x16 bf16 : lane = (n&15) + 16*(k>>4),     elem = k & 15
//   C/D 16x16 f32: lane = n (0..15 | 16..31),     vgpr v -> m = v + 8*(lane>>4)
// Fragments are stored pre-packed so each lane loads its 16 bf16 contiguously.

typedef __attribute__((ext_vector_type(16))) __bf16 bf16x16;
typedef __attribute__((ext_vector_type(8)))  float  f32x8;

#define VOCAB 128
#define EMBED 256
#define HID   1024
#define BATCH 64
#define SEQ   1024
#define H3    (3*HID)

__device__ __forceinline__ float sigmoidf_(float x) { return 1.0f / (1.0f + __expf(-x)); }

// ---------------------------------------------------------------------------
// eg[r][n] = b_in[n] + sum_k emb[r][k] * w_in[k][n]   (exact fp32, tiny GEMM)
// grid: 128 rows * 12 column blocks = 1536 blocks of 256
__global__ void eg_gemm(const float* __restrict__ emb, const float* __restrict__ w_in,
                        const float* __restrict__ b_in, float* __restrict__ eg) {
  const int r = blockIdx.x / 12;
  const int n = (blockIdx.x % 12) * 256 + threadIdx.x;
  const float* er = emb + r * EMBED;
  float s = b_in[n];
#pragma unroll 4
  for (int k = 0; k < EMBED; ++k) s += er[k] * w_in[k * H3 + n];
  eg[r * H3 + n] = s;
}

// ---------------------------------------------------------------------------
// Pack w_rec [HID][3*HID] fp32 -> bf16 B-fragments [g][ct][kt][lane][16]
// grid: 1024 rows * 12 blocks = 12288 blocks of 256
__global__ void pack_wrec(const float* __restrict__ w, __bf16* __restrict__ f) {
  const int k = blockIdx.x / 12;
  const int n = (blockIdx.x % 12) * 256 + threadIdx.x;
  const float v = w[k * H3 + n];
  const int g = n >> 10, c = n & 1023;
  const int ct = c >> 4, nn = c & 15;
  const int kt = k >> 5, kk = k & 31;
  const int lane = nn + ((kk >> 4) << 4);
  const int i = kk & 15;
  f[((((g * 64 + ct) * 32) + kt) << 9) + lane * 16 + i] = (__bf16)v;
}

// Pack w_out [HID][VOCAB] fp32 -> bf16 B-fragments [vt][kt][lane][16]
// grid: 131072 / 256 = 512 blocks
__global__ void pack_wout(const float* __restrict__ w, __bf16* __restrict__ f) {
  const int idx = blockIdx.x * 256 + threadIdx.x;
  const int k = idx >> 7, v = idx & 127;
  const float val = w[idx];
  const int vt = v >> 4, nn = v & 15;
  const int kt = k >> 5, kk = k & 31;
  const int lane = nn + ((kk >> 4) << 4);
  const int i = kk & 15;
  f[(((vt * 32) + kt) << 9) + lane * 16 + i] = (__bf16)val;
}

// Zero h state (fp32 + bf16 A-fragments). grid: 256 blocks of 256
__global__ void zero_h(float* __restrict__ h32, __bf16* __restrict__ hfrag) {
  const int idx = blockIdx.x * 256 + threadIdx.x;   // < 65536
  h32[idx] = 0.0f;
  hfrag[idx] = (__bf16)0.0f;
}

// ---------------------------------------------------------------------------
// One GRU timestep. grid: 64 blocks (16-column tiles of H) x 128 threads.
// Wave w owns M-tile w (batch rows 16w..16w+15), accumulates z/r/cand 16x16
// tiles over K=HID with v_wmma_f32_16x16x32_bf16, then fp32 gate math.
__global__ void __launch_bounds__(128)
gru_step(const int* __restrict__ ids, const float* __restrict__ eg,
         const __bf16* __restrict__ wrec_frag, const float* __restrict__ b_rec,
         float* __restrict__ h32, const __bf16* __restrict__ hfrag_in,
         __bf16* __restrict__ hfrag_out, __bf16* __restrict__ hs_frag, int t) {
  const int ct   = blockIdx.x;           // 16-wide column tile of H: 0..63
  const int wave = threadIdx.x >> 5;     // M-tile 0..3
  const int lane = threadIdx.x & 31;
  const int l    = lane & 15;
  const int hi   = lane >> 4;

  f32x8 az = {}, ar = {}, ac = {};
  const __bf16* ap = hfrag_in + ((wave * 32) << 9) + lane * 16;
  const __bf16* bz = wrec_frag + ((((0 * 64 + ct) * 32)) << 9) + lane * 16;
  const __bf16* br = wrec_frag + ((((1 * 64 + ct) * 32)) << 9) + lane * 16;
  const __bf16* bc = wrec_frag + ((((2 * 64 + ct) * 32)) << 9) + lane * 16;

#pragma unroll 2
  for (int kt = 0; kt < 32; ++kt) {
    bf16x16 a  = *(const bf16x16*)(ap + (kt << 9));
    bf16x16 vz = *(const bf16x16*)(bz + (kt << 9));
    bf16x16 vr = *(const bf16x16*)(br + (kt << 9));
    bf16x16 vc = *(const bf16x16*)(bc + (kt << 9));
    az = __builtin_amdgcn_wmma_f32_16x16x32_bf16(false, a, false, vz, (short)0, az, false, false);
    ar = __builtin_amdgcn_wmma_f32_16x16x32_bf16(false, a, false, vr, (short)0, ar, false, false);
    ac = __builtin_amdgcn_wmma_f32_16x16x32_bf16(false, a, false, vc, (short)0, ac, false, false);
  }

  const int n = ct * 16 + l;                        // hidden column (also K index)
  const float brz = b_rec[n];
  const float brr = b_rec[HID + n];
  const float brh = b_rec[2 * HID + n];
  const int kt2 = n >> 5, kk = n & 31;
  const int hi2 = kk >> 4;
  const int i2  = kk & 15;                          // A-fragment element index

#pragma unroll
  for (int v = 0; v < 8; ++v) {
    const int m  = wave * 16 + v + 8 * hi;          // batch row (C/D layout)
    const int id = ids[m * SEQ + t];
    const float* eb = eg + id * H3;
    const float xz = eb[n];
    const float xr = eb[HID + n];
    const float xh = eb[2 * HID + n];
    const float hold = h32[m * HID + n];
    const float z  = sigmoidf_(xz + az[v] + brz);
    const float r  = sigmoidf_(xr + ar[v] + brr);
    const float hh = tanhf(xh + r * (ac[v] + brh));
    const float hn = z * hold + (1.0f - z) * hh;
    h32[m * HID + n] = hn;                          // column-partitioned: race-free
    const int l2  = m & 15;
    const int dst = ((wave * 32 + kt2) << 9) + (l2 + 16 * hi2) * 16 + i2;
    const __bf16 hb = (__bf16)hn;
    hfrag_out[dst] = hb;                            // double-buffered vs. hfrag_in
    hs_frag[t * 65536 + dst] = hb;
  }
}

// ---------------------------------------------------------------------------
// logits = hs @ w_out + b_out. grid: 1024 blocks x 128; wave owns one (t, mt)
// M-tile and all 8 vocab N-tiles (A-fragment reused across 8 WMMAs per kstep).
__global__ void __launch_bounds__(128)
out_gemm(const __bf16* __restrict__ hs_frag, const __bf16* __restrict__ wout_frag,
         const float* __restrict__ b_out, float* __restrict__ logits) {
  const int wave = threadIdx.x >> 5;
  const int lane = threadIdx.x & 31;
  const int l = lane & 15, hi = lane >> 4;
  const int gt = blockIdx.x * 4 + wave;   // 0..4095
  const int t  = gt >> 2;
  const int mt = gt & 3;

  f32x8 acc[8];
#pragma unroll
  for (int i = 0; i < 8; ++i) acc[i] = (f32x8){};

  const __bf16* ap = hs_frag + t * 65536 + ((mt * 32) << 9) + lane * 16;
  const __bf16* bp = wout_frag + lane * 16;

  for (int kt = 0; kt < 32; ++kt) {
    bf16x16 a = *(const bf16x16*)(ap + (kt << 9));
#pragma unroll
    for (int vt = 0; vt < 8; ++vt) {
      bf16x16 b = *(const bf16x16*)(bp + (((vt * 32) + kt) << 9));
      acc[vt] = __builtin_amdgcn_wmma_f32_16x16x32_bf16(false, a, false, b, (short)0, acc[vt], false, false);
    }
  }

#pragma unroll
  for (int vt = 0; vt < 8; ++vt) {
    const int n = vt * 16 + l;
    const float bias = b_out[n];
#pragma unroll
    for (int v = 0; v < 8; ++v) {
      const int brow = mt * 16 + v + 8 * hi;
      logits[brow * (SEQ * VOCAB) + t * VOCAB + n] = acc[vt][v] + bias;
    }
  }
}

// ---------------------------------------------------------------------------
extern "C" void kernel_launch(void* const* d_in, const int* in_sizes, int n_in,
                              void* d_out, int out_size, void* d_ws, size_t ws_size,
                              hipStream_t stream) {
  const int*   ids   = (const int*)  d_in[0];
  const float* emb   = (const float*)d_in[1];
  const float* w_in  = (const float*)d_in[2];
  const float* b_in  = (const float*)d_in[3];
  const float* w_rec = (const float*)d_in[4];
  const float* b_rec = (const float*)d_in[5];
  const float* w_out = (const float*)d_in[6];
  const float* b_out = (const float*)d_in[7];
  float* logits = (float*)d_out;

  char* ws = (char*)d_ws;
  float*  eg     = (float*) (ws + 0);          //  1,572,864 B
  __bf16* wrecf  = (__bf16*)(ws + 1572864);    //  6,291,456 B
  __bf16* woutf  = (__bf16*)(ws + 7864320);    //    262,144 B
  float*  h32    = (float*) (ws + 8126464);    //    262,144 B
  __bf16* hfA    = (__bf16*)(ws + 8388608);    //    131,072 B
  __bf16* hfB    = (__bf16*)(ws + 8519680);    //    131,072 B
  __bf16* hsf    = (__bf16*)(ws + 8650752);    // 134,217,728 B  (total ~136 MB)

  eg_gemm  <<<1536,  256, 0, stream>>>(emb, w_in, b_in, eg);
  pack_wrec<<<12288, 256, 0, stream>>>(w_rec, wrecf);
  pack_wout<<<512,   256, 0, stream>>>(w_out, woutf);
  zero_h   <<<256,   256, 0, stream>>>(h32, hfA);

  __bf16* fin = hfA;
  __bf16* fout = hfB;
  for (int t = 0; t < SEQ; ++t) {
    gru_step<<<64, 128, 0, stream>>>(ids, eg, wrecf, b_rec, h32, fin, fout, hsf, t);
    __bf16* tmp = fin; fin = fout; fout = tmp;
  }

  out_gemm<<<1024, 128, 0, stream>>>(hsf, woutf, b_out, logits);
}
